// SimpleNodeModel_9440338117438
// MI455X (gfx1250) — compile-verified, matrix operand order checked by reading, not run on hardware
//
#include <hip/hip_runtime.h>
#include <hip/hip_bf16.h>
#include <math.h>

typedef __attribute__((ext_vector_type(16))) _Float16 v16h_t;
typedef __attribute__((ext_vector_type(8)))  _Float16 v8h_t;
typedef __attribute__((ext_vector_type(8)))  float    v8f_t;

#define PI_F 3.14159265358979f

// dims
#define ND   64
#define ED   32
#define GD   32
#define HD   128
#define DOF  6
#define K1P  192   // 2*ND+12+ED = 172 padded to 6*32
#define K3P  224   // HD+GD+ND = 224 = 7*32 exactly
#define N4P  80    // ND+DOF = 70 padded to 5*16
#define OUTD 77    // 64 + 9 + 3 + 1

// packed weight sizes (halfs)
#define W1PK (K1P*HD)   // 24576
#define W2PK (HD*HD)    // 16384
#define W3PK (K3P*HD)   // 28672
#define W4PK (HD*N4P)   // 10240
#define PKTOT (W1PK+W2PK+W3PK+W4PK)  // 79872 halfs -> 159744 bytes

#define WPB 4  // waves per block (wave32)

// ---------------------------------------------------------------------------
// WMMA helpers
// ---------------------------------------------------------------------------
__device__ inline v8f_t wmma16(v16h_t a, v16h_t b, v8f_t c) {
  return __builtin_amdgcn_wmma_f32_16x16x32_f16(false, a, false, b, (short)0, c,
                                                false, false);
}

// A fragment (16xK tile, K0..K0+31) from an LDS row-major f16 buffer.
// ISA 16-bit A layout: lanes 0-15 -> M, elems 0..7 = K0+half*8..+7,
// elems 8..15 = K0+half*8+16..+23.
__device__ inline v16h_t load_a_frag(const _Float16* feat, int ldk, int K0, int lane) {
  int m = lane & 15, half = lane >> 4;
  const _Float16* p = feat + m * ldk + K0 + half * 8;
  v8h_t lo = *(const v8h_t*)p;
  v8h_t hi = *(const v8h_t*)(p + 16);
  v16h_t a;
#pragma unroll
  for (int t = 0; t < 8; ++t) { a[t] = lo[t]; a[t + 8] = hi[t]; }
  return a;
}

// ---------------------------------------------------------------------------
// 3x3 helpers
// ---------------------------------------------------------------------------
__device__ inline void m3_mul(const float* A, const float* B, float* C) {  // C=A*B
#pragma unroll
  for (int r = 0; r < 3; ++r)
#pragma unroll
    for (int c = 0; c < 3; ++c)
      C[3 * r + c] = A[3 * r] * B[c] + A[3 * r + 1] * B[3 + c] + A[3 * r + 2] * B[6 + c];
}
__device__ inline void m3_mulT(const float* A, const float* B, float* C) { // C=A*B^T
#pragma unroll
  for (int r = 0; r < 3; ++r)
#pragma unroll
    for (int c = 0; c < 3; ++c)
      C[3 * r + c] = A[3 * r] * B[3 * c] + A[3 * r + 1] * B[3 * c + 1] + A[3 * r + 2] * B[3 * c + 2];
}
__device__ inline void m3_vec(const float* A, const float* x, float* y) {  // y=A*x
#pragma unroll
  for (int r = 0; r < 3; ++r)
    y[r] = A[3 * r] * x[0] + A[3 * r + 1] * x[1] + A[3 * r + 2] * x[2];
}

// ---------------------------------------------------------------------------
// Weight pack: fragment-native B layout. tile = kt*NT+nt; per tile 32 lanes x
// 16 halfs contiguous (one v16h per lane).
// ---------------------------------------------------------------------------
__device__ inline _Float16 pack_elem(const float* __restrict__ W, int K, int Nout,
                                     int NT, int idx) {
  int e = idx & 15;
  int lane = (idx >> 4) & 31;
  int tile = idx >> 9;
  int nt = tile % NT, kt = tile / NT;
  int n = nt * 16 + (lane & 15);
  int k = kt * 32 + ((lane >> 4) * 8) + (e < 8 ? e : e + 8);
  return (k < K && n < Nout) ? (_Float16)W[k * Nout + n] : (_Float16)0.f;
}

__global__ void pack_weights_kernel(const float* __restrict__ W1,
                                    const float* __restrict__ W2,
                                    const float* __restrict__ W3,
                                    const float* __restrict__ W4,
                                    _Float16* __restrict__ dst) {
  int idx = blockIdx.x * blockDim.x + threadIdx.x;
  if (idx >= PKTOT) return;
  if (idx < W1PK)
    dst[idx] = pack_elem(W1, 172, HD, 8, idx);
  else if (idx < W1PK + W2PK)
    dst[idx] = pack_elem(W2, HD, HD, 8, idx - W1PK);
  else if (idx < W1PK + W2PK + W3PK)
    dst[idx] = pack_elem(W3, K3P, HD, 8, idx - (W1PK + W2PK));
  else
    dst[idx] = pack_elem(W4, HD, ND + DOF, 5, idx - (W1PK + W2PK + W3PK));
}

__global__ void zero_accum_kernel(float* __restrict__ sums, float* __restrict__ cnt,
                                  int Nn) {
  int idx = blockIdx.x * blockDim.x + threadIdx.x;
  if (idx < Nn * HD) sums[idx] = 0.f;
  else if (idx < Nn * (HD + 1)) cnt[idx - Nn * HD] = 0.f;
}

// ---------------------------------------------------------------------------
// Edge kernel: rows [0,2E). row<E -> f_ij (target j), else f_ji (target i).
// Each wave32 owns 16 rows: build feature tile in its LDS slice, run
// L1 (192->128) and L2 (128->128) via WMMA (kt-outer so each A-fragment is
// read from LDS exactly once), ReLU, atomically scatter into per-node sums.
// ---------------------------------------------------------------------------
__global__ __launch_bounds__(WPB * 32) void edge_mlp_kernel(
    const float* __restrict__ xfeat, const float* __restrict__ T_R,
    const float* __restrict__ T_t, const float* __restrict__ edge_feat,
    const float* __restrict__ Tij_R, const float* __restrict__ Tij_t,
    const int* __restrict__ edge_index,
    const _Float16* __restrict__ W1pk, const float* __restrict__ b1,
    const _Float16* __restrict__ W2pk, const float* __restrict__ b2,
    float* __restrict__ sums, float* __restrict__ cnt, int Ecnt) {
  const int lane = threadIdx.x & 31;
  const int wv = threadIdx.x >> 5;
  const int E2 = 2 * Ecnt;
  __shared__ __align__(16) _Float16 featS[WPB][16][K1P];
  __shared__ __align__(16) _Float16 hS[WPB][16][HD];
  __shared__ int tgtS[WPB][16];
  const int base = (blockIdx.x * WPB + wv) * 16;

  // ---- phase 1a: per-row pose + target (lanes 0..15, one row each) ----
  if (lane < 16) {
    int row = base + lane;
    int tgt = -1;
    float v12[12];
#pragma unroll
    for (int k = 0; k < 12; ++k) v12[k] = 0.f;
    if (row < E2) {
      int d = row >= Ecnt ? 1 : 0;
      int e = d ? row - Ecnt : row;
      int i = edge_index[e];
      int j = edge_index[Ecnt + e];
      tgt = d ? i : j;
      float Ri[9], Rj[9], Th[9], ti[3], tj[3], tht[3];
#pragma unroll
      for (int k = 0; k < 9; ++k) {
        Ri[k] = T_R[i * 9 + k];
        Rj[k] = T_R[j * 9 + k];
        Th[k] = Tij_R[e * 9 + k];
      }
#pragma unroll
      for (int k = 0; k < 3; ++k) {
        ti[k] = T_t[i * 3 + k];
        tj[k] = T_t[j * 3 + k];
        tht[k] = Tij_t[e * 3 + k];
      }
      float Re[9], te[3], tmp[9], tv[3];
      if (!d) {
        // Ra = Rj Ri^T ; ta = tj - Ra ti ; Re = Ra Th^T ; te = ta - Re tht
        m3_mulT(Rj, Ri, tmp);
        m3_vec(tmp, ti, tv);
        float ta[3] = {tj[0] - tv[0], tj[1] - tv[1], tj[2] - tv[2]};
        m3_mulT(tmp, Th, Re);
        m3_vec(Re, tht, tv);
        te[0] = ta[0] - tv[0]; te[1] = ta[1] - tv[1]; te[2] = ta[2] - tv[2];
      } else {
        // Rb = Ri Rj^T ; tb = ti - Rb tj ; Re = Rb Th ; te = tb + Rb tht
        m3_mulT(Ri, Rj, tmp);
        m3_vec(tmp, tj, tv);
        float tb[3] = {ti[0] - tv[0], ti[1] - tv[1], ti[2] - tv[2]};
        m3_mul(tmp, Th, Re);
        m3_vec(tmp, tht, tv);
        te[0] = tb[0] + tv[0]; te[1] = tb[1] + tv[1]; te[2] = tb[2] + tv[2];
      }
#pragma unroll
      for (int r = 0; r < 3; ++r) {
        v12[4 * r + 0] = Re[3 * r + 0];
        v12[4 * r + 1] = Re[3 * r + 1];
        v12[4 * r + 2] = Re[3 * r + 2];
        v12[4 * r + 3] = te[r];
      }
      atomicAdd(&cnt[tgt], 1.0f);
    }
    tgtS[wv][lane] = tgt;
#pragma unroll
    for (int k = 0; k < 12; ++k)
      featS[wv][lane][160 + k] = (_Float16)v12[k];
#pragma unroll
    for (int k = 172; k < K1P; ++k) featS[wv][lane][k] = (_Float16)0.f;
  }

  // ---- phase 1b: gather xfeat/edge_feat (full wave, per row) ----
  for (int r = 0; r < 16; ++r) {
    int row = base + r;
    float f0 = 0.f, f1 = 0.f, f2 = 0.f, f3 = 0.f, f4 = 0.f;
    if (row < E2) {
      int d = row >= Ecnt ? 1 : 0;
      int e = d ? row - Ecnt : row;
      int i = edge_index[e];
      int j = edge_index[Ecnt + e];
      int a = d ? j : i;
      int b = d ? i : j;
      f0 = xfeat[a * ND + lane];
      f1 = xfeat[a * ND + 32 + lane];
      f2 = xfeat[b * ND + lane];
      f3 = xfeat[b * ND + 32 + lane];
      f4 = edge_feat[e * ED + lane];
    }
    featS[wv][r][lane]       = (_Float16)f0;
    featS[wv][r][32 + lane]  = (_Float16)f1;
    featS[wv][r][64 + lane]  = (_Float16)f2;
    featS[wv][r][96 + lane]  = (_Float16)f3;
    featS[wv][r][128 + lane] = (_Float16)f4;
  }

  const int half = lane >> 4;
  const int col = lane & 15;

  // ---- L1: 16x192 @ 192x128, +b1, ReLU -> hS (f16). kt-outer. ----
  {
    v8f_t acc[8];
#pragma unroll
    for (int nt = 0; nt < 8; ++nt) acc[nt] = (v8f_t){};
#pragma unroll
    for (int kt = 0; kt < 6; ++kt) {
      v16h_t a = load_a_frag(&featS[wv][0][0], K1P, kt * 32, lane);
#pragma unroll
      for (int nt = 0; nt < 8; ++nt) {
        v16h_t b = ((const v16h_t*)W1pk)[(kt * 8 + nt) * 32 + lane];
        acc[nt] = wmma16(a, b, acc[nt]);
      }
    }
#pragma unroll
    for (int nt = 0; nt < 8; ++nt) {
      int n = nt * 16 + col;
      float bb = b1[n];
#pragma unroll
      for (int v = 0; v < 8; ++v) {
        int m = v + 8 * half;
        hS[wv][m][n] = (_Float16)fmaxf(acc[nt][v] + bb, 0.f);
      }
    }
  }

  // ---- L2: 16x128 @ 128x128, +b2, ReLU, atomic scatter. kt-outer. ----
  {
    v8f_t acc[8];
#pragma unroll
    for (int nt = 0; nt < 8; ++nt) acc[nt] = (v8f_t){};
#pragma unroll
    for (int kt = 0; kt < 4; ++kt) {
      v16h_t a = load_a_frag(&hS[wv][0][0], HD, kt * 32, lane);
#pragma unroll
      for (int nt = 0; nt < 8; ++nt) {
        v16h_t b = ((const v16h_t*)W2pk)[(kt * 8 + nt) * 32 + lane];
        acc[nt] = wmma16(a, b, acc[nt]);
      }
    }
#pragma unroll
    for (int nt = 0; nt < 8; ++nt) {
      int n = nt * 16 + col;
      float bb = b2[n];
#pragma unroll
      for (int v = 0; v < 8; ++v) {
        int m = v + 8 * half;
        int tgt = tgtS[wv][m];
        if (tgt >= 0)
          atomicAdd(&sums[tgt * HD + n], fmaxf(acc[nt][v] + bb, 0.f));
      }
    }
  }
}

// ---------------------------------------------------------------------------
// Node kernel: g = [aggr(128) | xfeat(64) | u[batch](32)] (K=224) -> L3 -> L4
// -> SE(3) exp + compose epilogue. 16 nodes per wave.
// ---------------------------------------------------------------------------
__global__ __launch_bounds__(WPB * 32) void node_mlp_kernel(
    const float* __restrict__ xfeat, const float* __restrict__ T_R,
    const float* __restrict__ T_t, const float* __restrict__ u,
    const int* __restrict__ batch, const float* __restrict__ sums,
    const float* __restrict__ cnt,
    const _Float16* __restrict__ W3pk, const float* __restrict__ b3,
    const _Float16* __restrict__ W4pk, const float* __restrict__ b4,
    float* __restrict__ out, int Nn) {
  const int lane = threadIdx.x & 31;
  const int wv = threadIdx.x >> 5;
  __shared__ __align__(16) unsigned char smem[WPB * 16 * K3P * 2 + WPB * 16 * HD * 2];
  _Float16* featW = (_Float16*)(smem + wv * (16 * K3P * 2));          // 16 x 224
  _Float16* hW    = (_Float16*)(smem + WPB * 16 * K3P * 2 + wv * (16 * HD * 2));
  float*    outW  = (float*)(smem + wv * (16 * K3P * 2));             // aliases featW, 16 x 80
  const int base = (blockIdx.x * WPB + wv) * 16;

  // ---- phase 1: build g tile ----
  for (int r = 0; r < 16; ++r) {
    int node = base + r;
    float g0 = 0.f, g1 = 0.f, g2 = 0.f, g3 = 0.f, g4 = 0.f, g5 = 0.f, g6 = 0.f;
    if (node < Nn) {
      float inv = 1.f / fmaxf(cnt[node], 1.f);
      g0 = sums[node * HD + lane] * inv;
      g1 = sums[node * HD + 32 + lane] * inv;
      g2 = sums[node * HD + 64 + lane] * inv;
      g3 = sums[node * HD + 96 + lane] * inv;
      g4 = xfeat[node * ND + lane];
      g5 = xfeat[node * ND + 32 + lane];
      g6 = u[batch[node] * GD + lane];
    }
    featW[r * K3P + lane]        = (_Float16)g0;
    featW[r * K3P + 32 + lane]   = (_Float16)g1;
    featW[r * K3P + 64 + lane]   = (_Float16)g2;
    featW[r * K3P + 96 + lane]   = (_Float16)g3;
    featW[r * K3P + 128 + lane]  = (_Float16)g4;
    featW[r * K3P + 160 + lane]  = (_Float16)g5;
    featW[r * K3P + 192 + lane]  = (_Float16)g6;
  }

  const int half = lane >> 4;
  const int col = lane & 15;

  // ---- L3: 16x224 @ 224x128, +b3, ReLU -> hW (f16). kt-outer. ----
  {
    v8f_t acc[8];
#pragma unroll
    for (int nt = 0; nt < 8; ++nt) acc[nt] = (v8f_t){};
#pragma unroll
    for (int kt = 0; kt < 7; ++kt) {
      v16h_t a = load_a_frag(featW, K3P, kt * 32, lane);
#pragma unroll
      for (int nt = 0; nt < 8; ++nt) {
        v16h_t b = ((const v16h_t*)W3pk)[(kt * 8 + nt) * 32 + lane];
        acc[nt] = wmma16(a, b, acc[nt]);
      }
    }
#pragma unroll
    for (int nt = 0; nt < 8; ++nt) {
      int n = nt * 16 + col;
      float bb = b3[n];
#pragma unroll
      for (int v = 0; v < 8; ++v) {
        int m = v + 8 * half;
        hW[m * HD + n] = (_Float16)fmaxf(acc[nt][v] + bb, 0.f);
      }
    }
  }

  // ---- L4: 16x128 @ 128x80 (70 valid), +b4 (no ReLU) -> outW (f32, ld 80) ----
  {
    v8f_t acc[5];
#pragma unroll
    for (int nt = 0; nt < 5; ++nt) acc[nt] = (v8f_t){};
#pragma unroll
    for (int kt = 0; kt < 4; ++kt) {
      v16h_t a = load_a_frag(hW, HD, kt * 32, lane);
#pragma unroll
      for (int nt = 0; nt < 5; ++nt) {
        v16h_t b = ((const v16h_t*)W4pk)[(kt * 5 + nt) * 32 + lane];
        acc[nt] = wmma16(a, b, acc[nt]);
      }
    }
#pragma unroll
    for (int nt = 0; nt < 5; ++nt) {
      int n = nt * 16 + col;
      float bb = (n < ND + DOF) ? b4[n] : 0.f;
#pragma unroll
      for (int v = 0; v < 8; ++v) {
        int m = v + 8 * half;
        outW[m * N4P + n] = acc[nt][v] + bb;
      }
    }
  }

  // ---- epilogue: SE(3) exp + compose + write (lanes 0..15, one node each) ----
  if (lane < 16) {
    int node = base + lane;
    if (node < Nn) {
      const float* o = outW + lane * N4P;
      float rho[3] = {o[64], o[65], o[66]};
      float ph[3] = {o[67], o[68], o[69]};
      float th0 = sqrtf(ph[0] * ph[0] + ph[1] * ph[1] + ph[2] * ph[2]);
      float s = PI_F * tanhf(th0 / PI_F) / (th0 + 1e-8f);
      ph[0] *= s; ph[1] *= s; ph[2] *= s;
      float th2 = ph[0] * ph[0] + ph[1] * ph[1] + ph[2] * ph[2];
      float theta = sqrtf(th2);
      float th = sqrtf(th2 + 1e-12f);
      float ca, cb, cc;
      if (th < 1e-4f) {
        ca = 1.f - th2 / 6.f;
        cb = 0.5f - th2 / 24.f;
        cc = 1.f / 6.f - th2 / 120.f;
      } else {
        float sn = sinf(th), cs = cosf(th);
        ca = sn / th;
        cb = (1.f - cs) / th2;
        cc = (th - sn) / (th2 * th);
      }
      float Kx[9] = {0.f, -ph[2], ph[1], ph[2], 0.f, -ph[0], -ph[1], ph[0], 0.f};
      float KK[9];
      m3_mul(Kx, Kx, KK);
      float Rd[9], V[9];
#pragma unroll
      for (int k = 0; k < 9; ++k) {
        float I = (k == 0 || k == 4 || k == 8) ? 1.f : 0.f;
        Rd[k] = I + ca * Kx[k] + cb * KK[k];
        V[k]  = I + cb * Kx[k] + cc * KK[k];
      }
      float td[3];
      m3_vec(V, rho, td);
      float TR[9], Tt[3];
#pragma unroll
      for (int k = 0; k < 9; ++k) TR[k] = T_R[node * 9 + k];
#pragma unroll
      for (int k = 0; k < 3; ++k) Tt[k] = T_t[node * 3 + k];
      float Ro[9], tv[3];
      m3_mul(Rd, TR, Ro);
      m3_vec(Rd, Tt, tv);
      float to[3] = {tv[0] + td[0], tv[1] + td[1], tv[2] + td[2]};
      float* dst = out + node * OUTD;
#pragma unroll
      for (int c = 0; c < ND; ++c) dst[c] = xfeat[node * ND + c] + o[c];
#pragma unroll
      for (int k = 0; k < 9; ++k) dst[ND + k] = Ro[k];
      dst[73] = to[0]; dst[74] = to[1]; dst[75] = to[2];
      dst[76] = theta;
    }
  }
}

// ---------------------------------------------------------------------------
extern "C" void kernel_launch(void* const* d_in, const int* in_sizes, int n_in,
                              void* d_out, int out_size, void* d_ws, size_t ws_size,
                              hipStream_t stream) {
  const float* xfeat     = (const float*)d_in[0];
  const float* T_R       = (const float*)d_in[1];
  const float* T_t       = (const float*)d_in[2];
  const float* edge_feat = (const float*)d_in[3];
  const float* Tij_R     = (const float*)d_in[4];
  const float* Tij_t     = (const float*)d_in[5];
  const float* u         = (const float*)d_in[6];
  const int*   edge_index= (const int*)d_in[7];
  const int*   batch     = (const int*)d_in[8];
  const float* W1 = (const float*)d_in[9];
  const float* b1 = (const float*)d_in[10];
  const float* W2 = (const float*)d_in[11];
  const float* b2 = (const float*)d_in[12];
  const float* W3 = (const float*)d_in[13];
  const float* b3 = (const float*)d_in[14];
  const float* W4 = (const float*)d_in[15];
  const float* b4 = (const float*)d_in[16];
  float* out = (float*)d_out;

  const int Nn   = in_sizes[0] / ND;
  const int Ecnt = in_sizes[3] / ED;

  // workspace carve
  _Float16* Wpk = (_Float16*)d_ws;
  _Float16* W1pk = Wpk;
  _Float16* W2pk = W1pk + W1PK;
  _Float16* W3pk = W2pk + W2PK;
  _Float16* W4pk = W3pk + W3PK;
  float* sums = (float*)((char*)d_ws + (size_t)PKTOT * 2);  // 159744B, 256B-aligned
  float* cnt  = sums + (size_t)Nn * HD;

  // 1) pack weights to fragment-native f16
  pack_weights_kernel<<<(PKTOT + 255) / 256, 256, 0, stream>>>(W1, W2, W3, W4, Wpk);

  // 2) zero accumulators
  int ztot = Nn * (HD + 1);
  zero_accum_kernel<<<(ztot + 255) / 256, 256, 0, stream>>>(sums, cnt, Nn);

  // 3) edge MLP + scatter (2E rows, 64 rows per block)
  int rows = 2 * Ecnt;
  edge_mlp_kernel<<<(rows + WPB * 16 - 1) / (WPB * 16), WPB * 32, 0, stream>>>(
      xfeat, T_R, T_t, edge_feat, Tij_R, Tij_t, edge_index,
      W1pk, b1, W2pk, b2, sums, cnt, Ecnt);

  // 4) node MLP + SE(3) epilogue (64 nodes per block)
  node_mlp_kernel<<<(Nn + WPB * 16 - 1) / (WPB * 16), WPB * 32, 0, stream>>>(
      xfeat, T_R, T_t, u, batch, sums, cnt, W3pk, b3, W4pk, b4, out, Nn);
}
